// HANLayer_35296041238789
// MI455X (gfx1250) — matrix operand order repaired and checked by reading, not compiled, not run on hardware
//
#include <hip/hip_runtime.h>
#include <hip/hip_bf16.h>

// ---------------------------------------------------------------------------
// HAN layer on MI455X (gfx1250, wave32, WMMA + TDM).
// N=50000 nodes, E=800000 edges, P=3 metapaths, IN=128, H=8 heads, D=32, EMB=128
// ---------------------------------------------------------------------------

#define NN 50000
#define EE 800000
#define PP 3
#define INF_ 128
#define HH 8
#define DD 32
#define HD 256           // H*D
#define PHD 768          // P*H*D
#define EMB 128

typedef __attribute__((ext_vector_type(16))) _Float16 v16h;
typedef __attribute__((ext_vector_type(8)))  _Float16 v8h;
typedef __attribute__((ext_vector_type(4)))  _Float16 v4h;
typedef __attribute__((ext_vector_type(2)))  _Float16 v2h;
typedef __attribute__((ext_vector_type(8)))  float    v8f;
typedef __attribute__((ext_vector_type(4)))  unsigned u32x4;
typedef __attribute__((ext_vector_type(8)))  int      i32x8;
typedef __attribute__((ext_vector_type(4)))  int      i32x4;

// ---------------------------------------------------------------------------
// Tensor Data Mover support (gfx1250). Arity differs between toolchains:
// ROCm 7.2 clang-22 = 5 args, amdgpu-toolchain clang-23 (ships TDM header) = 6.
#if defined(__has_builtin)
#if __has_builtin(__builtin_amdgcn_tensor_load_to_lds) && \
    __has_builtin(__builtin_amdgcn_s_wait_tensorcnt)
#define HAVE_TDM 1
#endif
#endif
#if __has_include(<hip/amd_detail/amd_gfx1250_TDM.h>)
#define TDM_ARITY 6
#else
#define TDM_ARITY 5
#endif

#ifdef HAVE_TDM
// 2D f16 tile load: tile_w x tile_h elements, row stride `stride_elems`,
// into LDS at byte offset lds_addr.  D# layout per CDNA5 ISA ch.8.
__device__ __forceinline__ void tdm_load_2d_f16(unsigned lds_addr,
                                                const void* gptr,
                                                unsigned tile_w, unsigned tile_h,
                                                unsigned tensor_w, unsigned tensor_h,
                                                unsigned stride_elems) {
    unsigned long long ga = (unsigned long long)gptr;
    u32x4 g0;
    g0[0] = 1u;                                         // count=1 (valid user D#)
    g0[1] = lds_addr;                                   // lds_addr [63:32]
    g0[2] = (unsigned)(ga & 0xFFFFFFFFu);               // global_addr lo
    g0[3] = (unsigned)((ga >> 32) & 0x1FFFFFFu) | (2u << 30);  // addr hi | type=2
    i32x8 g1;
    g1[0] = (int)(1u << 16);                            // data_size=1 (2 bytes)
    g1[1] = (int)((tensor_w & 0xFFFFu) << 16);          // tensor_dim0[15:0]
    g1[2] = (int)(((tensor_w >> 16) & 0xFFFFu) | ((tensor_h & 0xFFFFu) << 16));
    g1[3] = (int)(((tensor_h >> 16) & 0xFFFFu) | ((tile_w & 0xFFFFu) << 16));
    g1[4] = (int)(tile_h & 0xFFFFu);                    // tile_dim1 | tile_dim2=0
    g1[5] = (int)stride_elems;                          // tensor_dim0_stride lo
    g1[6] = 0;                                          // stride0 hi | stride1 lo
    g1[7] = 0;                                          // stride1 hi
    i32x4 zz = {0, 0, 0, 0};
#if TDM_ARITY == 6
    i32x8 z8 = {0, 0, 0, 0, 0, 0, 0, 0};
    __builtin_amdgcn_tensor_load_to_lds(g0, g1, zz, zz, z8, 0);
#else
    __builtin_amdgcn_tensor_load_to_lds(g0, g1, zz, zz, 0);
#endif
}
#endif

// order-preserving float <-> uint map for atomicMax on signed floats
__device__ __forceinline__ unsigned mapf(float f) {
    unsigned u = __float_as_uint(f);
    return u ^ ((u >> 31) ? 0xFFFFFFFFu : 0x80000000u);
}
__device__ __forceinline__ float unmapf(unsigned u) {
    u ^= ((u >> 31) ? 0x80000000u : 0xFFFFFFFFu);
    return __uint_as_float(u);
}

// ---------------------------------------------------------------------------
__global__ void zero_kernel(float* __restrict__ p, int n) {
    int i = blockIdx.x * blockDim.x + threadIdx.x;
    if (i < n) p[i] = 0.0f;
}

// ---------------------------------------------------------------------------
// Wt[n*K + k] = (f16) W[k*Ncol + n]   (pre-transpose + convert, write-coalesced)
__global__ void w_transpose_kernel(const float* __restrict__ W,
                                   _Float16* __restrict__ Wt, int K, int Ncol) {
    int idx = blockIdx.x * blockDim.x + threadIdx.x;
    if (idx >= K * Ncol) return;
    int n = idx / K, k = idx % K;
    Wt[idx] = (_Float16)W[k * Ncol + n];
}

// ---------------------------------------------------------------------------
// feat = h (N x 128) @ W (128 x 256) via f16 WMMA, fp32 accumulate.
// 512 threads = 16 waves; wave w owns output cols [16w,16w+16).
// A (16x128) staged once; B^T staged per 32-wide K chunk from pre-converted Wt.
__global__ __launch_bounds__(512)
void feat_gemm_kernel(const float* __restrict__ h,
                      const _Float16* __restrict__ Wt,   // [256][128] n-major f16
                      float* __restrict__ feat) {
    __shared__ __align__(16) _Float16 As[16][128];   // 4 KB
    __shared__ __align__(16) _Float16 Bts[256][32];  // 16 KB

    const int tid   = threadIdx.x;
    const int wave  = tid >> 5;
    const int lane  = tid & 31;
    const int mbase = blockIdx.x * 16;     // 50000 = 3125*16
    const int nbase = wave * 16;
    const int row   = lane & 15;
    const int hf    = lane >> 4;

    // stage A once: 2048 f16, 512 threads * 4 (float4 load -> packed b64 store)
    {
        int r  = tid >> 5;
        int c4 = (tid & 31) * 4;
        float4 f4 = *(const float4*)&h[(mbase + r) * INF_ + c4];
        v4h a4 = {(_Float16)f4.x, (_Float16)f4.y, (_Float16)f4.z, (_Float16)f4.w};
        *(v4h*)&As[r][c4] = a4;
    }

    const int bn  = tid & 255;      // B^T row (output col)
    const int bc0 = (tid >> 8) * 16;

    v8f acc = {};
    for (int kk = 0; kk < 4; ++kk) {       // K = 128 = 4 * 32
        // stage B^T chunk: 2x global b128 + 2x ds b128 per thread
        v8h b0 = *(const v8h*)&Wt[bn * INF_ + kk * 32 + bc0];
        v8h b1 = *(const v8h*)&Wt[bn * INF_ + kk * 32 + bc0 + 8];
        *(v8h*)&Bts[bn][bc0]     = b0;
        *(v8h*)&Bts[bn][bc0 + 8] = b1;
        __syncthreads();

        const v8h* arow = (const v8h*)(&As[row][kk * 32]);
        v8h alo = arow[hf];
        v8h ahi = arow[2 + hf];
        const v8h* brow = (const v8h*)(&Bts[nbase + row][0]);
        v8h blo = brow[hf];
        v8h bhi = brow[2 + hf];
        v16h a, b;
        #pragma unroll
        for (int i = 0; i < 8; ++i) {
            a[i] = alo[i]; a[8 + i] = ahi[i];
            b[i] = blo[i]; b[8 + i] = bhi[i];
        }
        acc = __builtin_amdgcn_wmma_f32_16x16x32_f16(
                  false, a, false, b, (short)0, acc, false, false);
        __syncthreads();
    }
    #pragma unroll
    for (int v = 0; v < 8; ++v)
        feat[(mbase + v + 8 * hf) * HD + nbase + row] = acc[v];
}

// ---------------------------------------------------------------------------
// el[n,h] = feat[n,h,:] . attn_l[h,:] ; er likewise (float4 vectorized)
__global__ void attn_kernel(const float* __restrict__ feat,
                            const float* __restrict__ al,
                            const float* __restrict__ ar,
                            float* __restrict__ el,
                            float* __restrict__ er) {
    int idx = blockIdx.x * blockDim.x + threadIdx.x;
    if (idx >= NN * HH) return;
    int n = idx >> 3, hh = idx & 7;
    const float4* f = (const float4*)(feat + n * HD + hh * DD);
    const float4* a = (const float4*)(al + hh * DD);
    const float4* b = (const float4*)(ar + hh * DD);
    float sl = 0.f, sr = 0.f;
    #pragma unroll
    for (int d = 0; d < 8; ++d) {
        float4 fv = f[d], av = a[d], bv = b[d];
        sl += fv.x * av.x + fv.y * av.y + fv.z * av.z + fv.w * av.w;
        sr += fv.x * bv.x + fv.y * bv.y + fv.z * bv.z + fv.w * bv.w;
    }
    el[idx] = sl;
    er[idx] = sr;
}

// ---------------------------------------------------------------------------
// pass 1: e = leaky_relu(el[src]+er[dst]); store e; segment max by dst
__global__ void edge_max_kernel(const int* __restrict__ ei,
                                const float* __restrict__ el,
                                const float* __restrict__ er,
                                float* __restrict__ ebuf,
                                unsigned* __restrict__ mbuf) {
    int idx = blockIdx.x * blockDim.x + threadIdx.x;   // e*8 + h
    if (idx >= EE * HH) return;
    int e = idx >> 3, hh = idx & 7;
    int src = ei[e], dst = ei[EE + e];
    float v = el[src * HH + hh] + er[dst * HH + hh];
    v = (v > 0.f) ? v : 0.2f * v;
    ebuf[idx] = v;
    atomicMax(&mbuf[dst * HH + hh], mapf(v));
}

// pass 2: ex = exp(e - m[dst]); store ex; segment-sum denominators
__global__ void edge_exp_kernel(const int* __restrict__ ei,
                                float* __restrict__ ebuf,
                                const unsigned* __restrict__ mbuf,
                                float* __restrict__ den) {
    int idx = blockIdx.x * blockDim.x + threadIdx.x;
    if (idx >= EE * HH) return;
    int e = idx >> 3, hh = idx & 7;
    int dst = ei[EE + e];
    float ex = expf(ebuf[idx] - unmapf(mbuf[dst * HH + hh]));
    ebuf[idx] = ex;
    atomicAdd(&den[dst * HH + hh], ex);
}

// pass 3: agg[dst] += feat[src] * (ex/den[dst]);  4 edges/block, float4 loads
__global__ __launch_bounds__(256)
void msg_kernel(const int* __restrict__ ei,
                const float* __restrict__ feat,
                const float* __restrict__ ebuf,
                const float* __restrict__ den,
                float* __restrict__ agg) {
    int e  = blockIdx.x * 4 + (threadIdx.x >> 6);
    int t  = threadIdx.x & 63;          // 64 lanes/edge, 4 floats each
    int hh = t >> 3;                    // (4t)/32
    int src = ei[e], dst = ei[EE + e];
    float a = ebuf[e * HH + hh] / den[dst * HH + hh];
    float4 f = *(const float4*)&feat[src * HD + t * 4];
    float* o = &agg[dst * HD + t * 4];
    atomicAdd(o + 0, f.x * a);
    atomicAdd(o + 1, f.y * a);
    atomicAdd(o + 2, f.z * a);
    atomicAdd(o + 3, f.w * a);
}

// ---------------------------------------------------------------------------
// z[n, p*256 + c] = f16(elu(agg[n,c]))  (packed v2h stores)
__global__ __launch_bounds__(128)
void elu_pack_kernel(const float* __restrict__ agg,
                     _Float16* __restrict__ z, int p) {
    int n  = blockIdx.x;
    int c2 = threadIdx.x * 2;
    float v0 = agg[n * HD + c2];
    float v1 = agg[n * HD + c2 + 1];
    v0 = (v0 > 0.f) ? v0 : (expf(v0) - 1.0f);
    v1 = (v1 > 0.f) ? v1 : (expf(v1) - 1.0f);
    v2h pk = {(_Float16)v0, (_Float16)v1};
    *(v2h*)&z[n * PHD + p * HD + c2] = pk;
}

// ---------------------------------------------------------------------------
// out = z (N x 768, f16) @ sem_w (768 x 128) + sem_b ; 8 waves per block.
// Whole A tile (16x768 f16 = 24 KB) staged once -- via TDM when available.
__global__ __launch_bounds__(256)
void sem_gemm_kernel(const _Float16* __restrict__ z,
                     const _Float16* __restrict__ swt,  // [128][768] n-major f16
                     const float* __restrict__ sb,
                     float* __restrict__ out) {
    __shared__ __align__(16) _Float16 As[16][PHD];   // 24 KB
    __shared__ __align__(16) _Float16 Bts[128][32];  // 8 KB

    const int tid   = threadIdx.x;
    const int wave  = tid >> 5;
    const int lane  = tid & 31;
    const int mbase = blockIdx.x * 16;
    const int nbase = wave * 16;           // 128 = 8 waves * 16
    const int row   = lane & 15;
    const int hf    = lane >> 4;

#ifdef HAVE_TDM
    if (wave == 0) {
        unsigned lds_off = (unsigned)(unsigned long long)(&As[0][0]);
        tdm_load_2d_f16(lds_off, z + (size_t)mbase * PHD,
                        /*tile*/ PHD, 16,
                        /*tensor*/ PHD, (unsigned)(NN - mbase),
                        /*stride*/ PHD);
        __builtin_amdgcn_s_wait_tensorcnt(0);
    }
#else
    #pragma unroll
    for (int j = 0; j < 6; ++j) {          // 1536 v8h chunks / 256 threads
        int c8 = tid + j * 256;
        int r = c8 / 96, col = (c8 % 96) * 8;
        v8h v = *(const v8h*)&z[(size_t)(mbase + r) * PHD + col];
        *(v8h*)&As[r][col] = v;
    }
#endif

    const int bn  = tid & 127;
    const int bc0 = (tid >> 7) * 16;

    v8f acc = {};
    for (int kk = 0; kk < 24; ++kk) {      // K = 768 = 24 * 32
        v8h b0 = *(const v8h*)&swt[bn * PHD + kk * 32 + bc0];
        v8h b1 = *(const v8h*)&swt[bn * PHD + kk * 32 + bc0 + 8];
        *(v8h*)&Bts[bn][bc0]     = b0;
        *(v8h*)&Bts[bn][bc0 + 8] = b1;
        __syncthreads();

        const v8h* arow = (const v8h*)(&As[row][kk * 32]);
        v8h alo = arow[hf];
        v8h ahi = arow[2 + hf];
        const v8h* brow = (const v8h*)(&Bts[nbase + row][0]);
        v8h blo = brow[hf];
        v8h bhi = brow[2 + hf];
        v16h a, b;
        #pragma unroll
        for (int i = 0; i < 8; ++i) {
            a[i] = alo[i]; a[8 + i] = ahi[i];
            b[i] = blo[i]; b[8 + i] = bhi[i];
        }
        acc = __builtin_amdgcn_wmma_f32_16x16x32_f16(
                  false, a, false, b, (short)0, acc, false, false);
        __syncthreads();
    }
    float bias = sb[nbase + row];
    #pragma unroll
    for (int v = 0; v < 8; ++v)
        out[(mbase + v + 8 * hf) * EMB + nbase + row] = acc[v] + bias;
}

// ---------------------------------------------------------------------------
extern "C" void kernel_launch(void* const* d_in, const int* in_sizes, int n_in,
                              void* d_out, int out_size, void* d_ws, size_t ws_size,
                              hipStream_t stream) {
    const float* h      = (const float*)d_in[0];
    const int*   ei     = (const int*)d_in[1];     // (P, 2, E)
    const float* fc_w   = (const float*)d_in[2];   // (P, 128, 256)
    const float* attn_l = (const float*)d_in[3];   // (P, 8, 32)
    const float* attn_r = (const float*)d_in[4];
    const float* sem_w  = (const float*)d_in[5];   // (768, 128)
    const float* sem_b  = (const float*)d_in[6];   // (128,)
    float* out = (float*)d_out;

    // workspace carve-up (per-p buffers reused across metapaths)
    char* w = (char*)d_ws;
    float*    feat = (float*)w;      w += (size_t)NN * HD * 4;      // 51.2 MB
    float*    agg  = (float*)w;      w += (size_t)NN * HD * 4;      // 51.2 MB
    float*    ebuf = (float*)w;      w += (size_t)EE * HH * 4;      // 25.6 MB
    float*    el   = (float*)w;      w += (size_t)NN * HH * 4;
    float*    er   = (float*)w;      w += (size_t)NN * HH * 4;
    unsigned* mbuf = (unsigned*)w;   w += (size_t)NN * HH * 4;
    float*    den  = (float*)w;      w += (size_t)NN * HH * 4;
    _Float16* z    = (_Float16*)w;   w += (size_t)NN * PHD * 2;     // 76.8 MB
    _Float16* Wth  = (_Float16*)w;   w += (size_t)PP * HD * INF_ * 2;
    _Float16* swt  = (_Float16*)w;   w += (size_t)EMB * PHD * 2;

    const int ZB = 256;

    // pre-transpose + f16-convert weights (tiny, once per call)
    for (int p = 0; p < PP; ++p)
        w_transpose_kernel<<<(INF_ * HD + ZB - 1) / ZB, ZB, 0, stream>>>(
            fc_w + (size_t)p * INF_ * HD, Wth + (size_t)p * HD * INF_, INF_, HD);
    w_transpose_kernel<<<(PHD * EMB + ZB - 1) / ZB, ZB, 0, stream>>>(
        sem_w, swt, PHD, EMB);

    for (int p = 0; p < PP; ++p) {
        const int*      ei_p = ei + (size_t)p * 2 * EE;
        const _Float16* Wt_p = Wth + (size_t)p * HD * INF_;
        const float*    al_p = attn_l + (size_t)p * HH * DD;
        const float*    ar_p = attn_r + (size_t)p * HH * DD;

        zero_kernel<<<(NN * HD + ZB - 1) / ZB, ZB, 0, stream>>>(agg, NN * HD);
        zero_kernel<<<(NN * HH + ZB - 1) / ZB, ZB, 0, stream>>>(den, NN * HH);
        zero_kernel<<<(NN * HH + ZB - 1) / ZB, ZB, 0, stream>>>((float*)mbuf, NN * HH);

        feat_gemm_kernel<<<NN / 16, 512, 0, stream>>>(h, Wt_p, feat);
        attn_kernel<<<(NN * HH + 255) / 256, 256, 0, stream>>>(feat, al_p, ar_p, el, er);
        edge_max_kernel<<<(EE * HH) / 256, 256, 0, stream>>>(ei_p, el, er, ebuf, mbuf);
        edge_exp_kernel<<<(EE * HH) / 256, 256, 0, stream>>>(ei_p, ebuf, mbuf, den);
        msg_kernel<<<EE / 4, 256, 0, stream>>>(ei_p, feat, ebuf, den, agg);
        elu_pack_kernel<<<NN, 128, 0, stream>>>(agg, z, p);
    }
    sem_gemm_kernel<<<NN / 16, 256, 0, stream>>>(z, swt, sem_b, out);
}
